// StackedLSTM_47399259079386
// MI455X (gfx1250) — compile-verified, matrix operand order checked by reading, not run on hardware
//
#include <hip/hip_runtime.h>
#include <math.h>

// ---------------------------------------------------------------------------
// StackedLSTM for MI455X (gfx1250, wave32, WMMA f32_16x16x32_f16)
//   T=512, B=64, H=1024, L=4, dtype f16
//
// Per layer:
//   1) transpose W_l, U_l ([H,4H] -> [4H,H]) so B-fragments are contiguous
//   2) G = X @ W_l + b : LDS-tiled WMMA GEMM, A/B panels staged with
//      global_load_async_to_lds_b128 (ASYNCcnt), double-buffered
//   3) persistent recurrent kernel: 512 steps, acc = G_t + h @ U_l,
//      4 gate tiles per wave -> pointwise LSTM math fully in-register,
//      device-wide barrier per timestep
// ---------------------------------------------------------------------------

typedef __attribute__((ext_vector_type(16))) _Float16 v16h;
typedef __attribute__((ext_vector_type(8)))  _Float16 v8h;
typedef __attribute__((ext_vector_type(8)))  float    v8f;

#define T_  512
#define B_  64
#define H_  1024
#define L_  4
#define G4_ (4 * H_)   // 4096

// LDS panel geometry for the big GEMM: 128 rows x 32 halves, padded row
// stride of 40 halves (80 B = 16B-aligned; row*20 words mod 64 banks is
// conflict-free across 16 consecutive rows).
#define PAN_ROWS 128
#define PAN_STRH 40            // halves per row in LDS
#define PAN_STRB (PAN_STRH*2)  // bytes per row in LDS

// Per-lane 16-bit WMMA fragment for one 16x32 tile: element j of the v16h is
// K = 8*hi + j (j<8) and 16 + 8*hi + (j-8): two contiguous 16-byte runs.
__device__ __forceinline__ v16h load_frag16(const _Float16* __restrict__ p, int hi) {
  const v8h lo = *(const v8h*)(p + 8 * hi);
  const v8h hh = *(const v8h*)(p + 16 + 8 * hi);
  v16h r;
#pragma unroll
  for (int i = 0; i < 8; ++i) { r[i] = lo[i]; r[i + 8] = hh[i]; }
  return r;
}

__device__ __forceinline__ float sigmoidf_(float x) {
  return 1.0f / (1.0f + __expf(-x));
}

// Async DMA: global -> LDS, 16 bytes, tracked by ASYNCcnt (no VGPR data).
__device__ __forceinline__ void async_ld_b128(unsigned lds_addr, const _Float16* g) {
  asm volatile("global_load_async_to_lds_b128 %0, %1, off"
               :: "v"(lds_addr), "v"(g)
               : "memory");
}
__device__ __forceinline__ void wait_async_le4() {
  asm volatile("s_wait_asynccnt 0x4" ::: "memory");
}
__device__ __forceinline__ void wait_async_0() {
  asm volatile("s_wait_asynccnt 0x0" ::: "memory");
}

// ---------------------------------------------------------------------------
// Transpose [H_ x G4_] f16 row-major -> [G4_ x H_] row-major. LDS-tiled.
// grid: (G4_/32, H_/32), block: (32, 8)
// ---------------------------------------------------------------------------
__global__ __launch_bounds__(256) void lstm_transpose_k(
    const _Float16* __restrict__ in,   // [H_, G4_]
    _Float16* __restrict__ out)        // [G4_, H_]
{
  __shared__ _Float16 tile[32][33];
  const int tx = threadIdx.x;
  const int ty = threadIdx.y;
  const int n0 = blockIdx.x * 32;
  const int k0 = blockIdx.y * 32;
#pragma unroll
  for (int j = 0; j < 32; j += 8)
    tile[ty + j][tx] = in[(size_t)(k0 + ty + j) * G4_ + (n0 + tx)];
  __syncthreads();
#pragma unroll
  for (int j = 0; j < 32; j += 8)
    out[(size_t)(n0 + ty + j) * H_ + (k0 + tx)] = tile[tx][ty + j];
}

// ---------------------------------------------------------------------------
// G = X @ W + bias   (M = T_*B_ = 32768, K = H_, N = G4_), f16 in, f32 acc.
// Workgroup: 256 threads (8 waves) computes a 128(M) x 128(N) tile.
// Wave (wm = w&3, wn = w>>2) computes 32x64 -> 2x4 WMMA accumulators.
// A/B 128x32 panels double-buffered in LDS via async global->LDS b128 DMA.
// grid = (M/128)*(N/128) = 8192 blocks.
// ---------------------------------------------------------------------------
__global__ __launch_bounds__(256) void lstm_gemm_xw_k(
    const _Float16* __restrict__ X,    // [M, H_]
    const _Float16* __restrict__ Wt,   // [G4_, H_]  (transposed W)
    const _Float16* __restrict__ bias, // [G4_]
    _Float16* __restrict__ G)          // [M, G4_]
{
  __shared__ _Float16 Alds[2][PAN_ROWS * PAN_STRH];
  __shared__ _Float16 Blds[2][PAN_ROWS * PAN_STRH];

  const int tid  = threadIdx.x;
  const int lane = tid & 31;
  const int wave = tid >> 5;
  const int m15  = lane & 15;
  const int hi   = lane >> 4;
  const int wm   = wave & 3;   // 0..3 -> 32-row sub-block
  const int wn   = wave >> 2;  // 0..1 -> 64-col sub-block

  const int NBN   = G4_ / 128;            // 32
  const int Mbase = (blockIdx.x / NBN) * 128;
  const int Nbase = (blockIdx.x % NBN) * 128;

  // staging assignment: thread copies 32B of one panel row (2 b128 each for A,B)
  const int srow  = tid >> 1;             // 0..127
  const int spart = tid & 1;              // which 16-half chunk of the row
  const _Float16* gA = X  + (size_t)(Mbase + srow) * H_ + spart * 16;
  const _Float16* gB = Wt + (size_t)(Nbase + srow) * H_ + spart * 16;
  const unsigned ldsA[2] = {
      (unsigned)(uintptr_t)&Alds[0][0] + (unsigned)(srow * PAN_STRB + spart * 32),
      (unsigned)(uintptr_t)&Alds[1][0] + (unsigned)(srow * PAN_STRB + spart * 32)};
  const unsigned ldsB[2] = {
      (unsigned)(uintptr_t)&Blds[0][0] + (unsigned)(srow * PAN_STRB + spart * 32),
      (unsigned)(uintptr_t)&Blds[1][0] + (unsigned)(srow * PAN_STRB + spart * 32)};

  v8f acc[2][4];
#pragma unroll
  for (int i = 0; i < 2; ++i)
#pragma unroll
    for (int j = 0; j < 4; ++j)
      acc[i][j] = (v8f){0.f, 0.f, 0.f, 0.f, 0.f, 0.f, 0.f, 0.f};

  const int NK = H_ / 32;   // 32 k-tiles

  // prologue: stage k-tile 0 into buffer 0
  async_ld_b128(ldsA[0],      gA);
  async_ld_b128(ldsA[0] + 16, gA + 8);
  async_ld_b128(ldsB[0],      gB);
  async_ld_b128(ldsB[0] + 16, gB + 8);

  for (int kt = 0; kt < NK; ++kt) {
    const int cur = kt & 1;
    if (kt + 1 < NK) {
      const int nxt = cur ^ 1;
      const _Float16* ga = gA + (kt + 1) * 32;
      const _Float16* gb = gB + (kt + 1) * 32;
      async_ld_b128(ldsA[nxt],      ga);
      async_ld_b128(ldsA[nxt] + 16, ga + 8);
      async_ld_b128(ldsB[nxt],      gb);
      async_ld_b128(ldsB[nxt] + 16, gb + 8);
      wait_async_le4();   // current tile's 4 DMAs done; next 4 in flight
    } else {
      wait_async_0();
    }
    __syncthreads();

    const _Float16* ap = &Alds[cur][0];
    const _Float16* bp = &Blds[cur][0];
    v16h afr[2];
#pragma unroll
    for (int i = 0; i < 2; ++i)
      afr[i] = load_frag16(ap + (wm * 32 + i * 16 + m15) * PAN_STRH, hi);
#pragma unroll
    for (int j = 0; j < 4; ++j) {
      const v16h bf = load_frag16(bp + (wn * 64 + j * 16 + m15) * PAN_STRH, hi);
      acc[0][j] = __builtin_amdgcn_wmma_f32_16x16x32_f16(
          false, afr[0], false, bf, (short)0, acc[0][j], false, false);
      acc[1][j] = __builtin_amdgcn_wmma_f32_16x16x32_f16(
          false, afr[1], false, bf, (short)0, acc[1][j], false, false);
    }
    __syncthreads();  // reads done before this buffer is DMA-overwritten
  }

  // epilogue: + bias, f16 store. C/D layout: n = lane&15, m = v + 8*hi
#pragma unroll
  for (int j = 0; j < 4; ++j) {
    const int col = Nbase + wn * 64 + j * 16 + m15;
    const float bv = (float)bias[col];
#pragma unroll
    for (int i = 0; i < 2; ++i) {
#pragma unroll
      for (int v = 0; v < 8; ++v) {
        const int row = Mbase + wm * 32 + i * 16 + v + 8 * hi;
        G[(size_t)row * G4_ + col] = (_Float16)(acc[i][j][v] + bv);
      }
    }
  }
}

// ---------------------------------------------------------------------------
// Persistent recurrent kernel (one layer): 512 timesteps with a device-wide
// barrier per step. grid = H_/16 = 64 blocks, block = 128 (4 waves).
// Block bx owns hidden columns [bx*16, bx*16+16); wave w owns batch rows
// [16w, 16w+16) and ALL FOUR gate tiles -> pointwise LSTM math in-register;
// cell state c lives in registers across all 512 steps.
// ---------------------------------------------------------------------------
__global__ __launch_bounds__(128) void lstm_recurrent_k(
    const _Float16* __restrict__ G,      // [T_*B_, G4_]  (xW + b)
    const _Float16* __restrict__ Ut,     // [G4_, H_]     (transposed U)
    _Float16* __restrict__ hbuf,         // [2][B_, H_]   ping-pong h
    unsigned* __restrict__ barrier_cnt,  // zeroed before launch
    _Float16* __restrict__ hs,           // [T_, B_, H_] out
    _Float16* __restrict__ cs)           // [T_, B_, H_] out
{
  const int lane = threadIdx.x & 31;
  const int wave = threadIdx.x >> 5;     // batch rows 16w..16w+15
  const int m15  = lane & 15;
  const int hi   = lane >> 4;
  const int n0   = blockIdx.x * 16;      // hidden-column block
  const int col  = n0 + m15;

  const _Float16* brow[4];
#pragma unroll
  for (int q = 0; q < 4; ++q)
    brow[q] = Ut + (size_t)(q * H_ + n0 + m15) * H_;

  float cprev[8];
#pragma unroll
  for (int v = 0; v < 8; ++v) cprev[v] = 0.0f;

  for (int t = 0; t < T_; ++t) {
    const _Float16* hcur = hbuf + (size_t)(t & 1) * (B_ * H_);
    _Float16*       hnxt = hbuf + (size_t)((t + 1) & 1) * (B_ * H_);

    // init accumulators with precomputed xW + b
    v8f acc[4];
    const _Float16* gbase = G + ((size_t)t * B_ + wave * 16) * G4_;
#pragma unroll
    for (int q = 0; q < 4; ++q) {
#pragma unroll
      for (int v = 0; v < 8; ++v) {
        const int m = v + 8 * hi;
        acc[q][v] = (float)gbase[(size_t)m * G4_ + q * H_ + n0 + m15];
      }
    }

    // prefetch next timestep's G rows (global_prefetch_b8) while we K-loop
    if (t + 1 < T_) {
      const _Float16* gnext = G + ((size_t)(t + 1) * B_ + wave * 16) * G4_;
#pragma unroll
      for (int q = 0; q < 4; ++q)
        __builtin_prefetch(gnext + (size_t)(8 * hi) * G4_ + q * H_ + n0 + m15, 0, 1);
    }

    // acc += h_{t-1} @ U   (K = H_ in steps of 32)
    const _Float16* arow = hcur + (size_t)(wave * 16 + m15) * H_;
    for (int k0 = 0; k0 < H_; k0 += 32) {
      const v16h a = load_frag16(arow + k0, hi);
#pragma unroll
      for (int q = 0; q < 4; ++q) {
        const v16h b = load_frag16(brow[q] + k0, hi);
        acc[q] = __builtin_amdgcn_wmma_f32_16x16x32_f16(
            false, a, false, b, (short)0, acc[q], false, false);
      }
    }

    // gates + state update, in-register; f16 rounding matches reference carry
#pragma unroll
    for (int v = 0; v < 8; ++v) {
      const float ig = sigmoidf_(acc[0][v]);
      const float fg = sigmoidf_(acc[1][v]);
      const float og = sigmoidf_(acc[2][v]);
      const float gc = tanhf(acc[3][v]);
      float cn = fg * cprev[v] + ig * gc;
      const _Float16 c16 = (_Float16)cn;
      cn = (float)c16;
      cprev[v] = cn;
      const _Float16 h16 = (_Float16)(og * tanhf(cn));

      const int m = v + 8 * hi;
      const int b = wave * 16 + m;
      const size_t oidx = ((size_t)t * B_ + b) * H_ + col;
      hs[oidx] = h16;
      cs[oidx] = c16;
      hnxt[(size_t)b * H_ + col] = h16;
    }

    // device-wide barrier: publish h_t, then monotonic-counter rendezvous
    __threadfence();
    __syncthreads();
    if (threadIdx.x == 0) {
      __hip_atomic_fetch_add(barrier_cnt, 1u, __ATOMIC_ACQ_REL,
                             __HIP_MEMORY_SCOPE_AGENT);
      const unsigned target = (unsigned)gridDim.x * (unsigned)(t + 1);
      while (__hip_atomic_load(barrier_cnt, __ATOMIC_ACQUIRE,
                               __HIP_MEMORY_SCOPE_AGENT) < target) {
        __builtin_amdgcn_s_sleep(2);
      }
    }
    __syncthreads();
  }
}

// ---------------------------------------------------------------------------
// Host-side orchestration (graph-capture safe: kernel launches and
// hipMemsetAsync on `stream` only).
// ---------------------------------------------------------------------------
extern "C" void kernel_launch(void* const* d_in, const int* in_sizes, int n_in,
                              void* d_out, int out_size, void* d_ws, size_t ws_size,
                              hipStream_t stream) {
  (void)in_sizes; (void)n_in; (void)out_size; (void)ws_size;

  const _Float16* xs   = (const _Float16*)d_in[0];  // [T,B,H]
  const _Float16* W    = (const _Float16*)d_in[1];  // [L,H,4H]
  const _Float16* U    = (const _Float16*)d_in[2];  // [L,H,4H]
  const _Float16* bias = (const _Float16*)d_in[3];  // [L,4H]

  _Float16* out = (_Float16*)d_out;
  const size_t THB = (size_t)T_ * B_ * H_;
  _Float16* hiddens = out;                       // [L,T,B,H]
  _Float16* cells   = out + (size_t)L_ * THB;    // [L,T,B,H]

  char* ws = (char*)d_ws;
  size_t off = 0;
  _Float16* Gbuf = (_Float16*)(ws + off); off += (size_t)T_ * B_ * G4_ * sizeof(_Float16); // 256 MB
  _Float16* Wt   = (_Float16*)(ws + off); off += (size_t)G4_ * H_ * sizeof(_Float16);      // 8 MB
  _Float16* Ut   = (_Float16*)(ws + off); off += (size_t)G4_ * H_ * sizeof(_Float16);      // 8 MB
  _Float16* hbuf = (_Float16*)(ws + off); off += (size_t)2 * B_ * H_ * sizeof(_Float16);   // 256 KB
  unsigned* cnt  = (unsigned*)(ws + off);  off += 256;

  const dim3 tgrid(G4_ / 32, H_ / 32);
  const dim3 tblk(32, 8);
  const int  gemm_blocks = ((T_ * B_) / 128) * (G4_ / 128);  // 8192

  for (int l = 0; l < L_; ++l) {
    const _Float16* Xl = (l == 0) ? xs : (hiddens + (size_t)(l - 1) * THB);
    const _Float16* Wl = W + (size_t)l * H_ * G4_;
    const _Float16* Ul = U + (size_t)l * H_ * G4_;
    const _Float16* bl = bias + (size_t)l * G4_;

    lstm_transpose_k<<<tgrid, tblk, 0, stream>>>(Wl, Wt);
    lstm_transpose_k<<<tgrid, tblk, 0, stream>>>(Ul, Ut);
    lstm_gemm_xw_k<<<gemm_blocks, 256, 0, stream>>>(Xl, Wt, bl, Gbuf);

    hipMemsetAsync(hbuf, 0, (size_t)2 * B_ * H_ * sizeof(_Float16), stream);
    hipMemsetAsync(cnt, 0, sizeof(unsigned), stream);

    lstm_recurrent_k<<<H_ / 16, 128, 0, stream>>>(
        Gbuf, Ut, hbuf, cnt,
        hiddens + (size_t)l * THB, cells + (size_t)l * THB);
  }
}
